// LSTMLanguageModel_78941498901138
// MI455X (gfx1250) — compile-verified, matrix operand order checked by reading, not run on hardware
//
#include <hip/hip_runtime.h>
#include <hip/hip_bf16.h>
#include <stdint.h>

// ---------------------------------------------------------------------------
// LSTM language model forward for MI455X (gfx1250, wave32, WMMA).
// GEMMs: v_wmma_f32_16x16x32_bf16, LDS-staged B tiles via async-to-LDS,
// software-pipelined K loops, TDM (tensor_load_to_lds) for Whh staging.
// ---------------------------------------------------------------------------

typedef __bf16 bf16_t;
typedef __attribute__((ext_vector_type(16))) __bf16 v16bf;
typedef __attribute__((ext_vector_type(8)))  __bf16 v8bf;
typedef __attribute__((ext_vector_type(8)))  float  v8f;

// Exact pointer types for the async-to-LDS builtin (from hipcc diagnostics:
// param 0 is "vector_size(16) int __device__ *", i.e. v4i in addrspace(1)).
typedef int v4i_vs __attribute__((vector_size(16)));
typedef v4i_vs __attribute__((address_space(1)))* as1_v4i_t;
typedef v4i_vs __attribute__((address_space(3)))* as3_v4i_t;

// TDM descriptor group types (6-arg toolchain form:
//  (u32x4 g0, i32x8 g1, i32x4 g2, i32x4 g3, i32x8 g4, i32 cpol)).
typedef unsigned u32x4_t __attribute__((vector_size(16)));
typedef int      i32x8_t __attribute__((vector_size(32)));
typedef int      i32x4_t __attribute__((vector_size(16)));

#define V_SZ 32000
#define E_SZ 1024
#define H_SZ 1024
#define B_SZ 16
#define T_SZ 256
#define G_SZ 4096   // 4*H (gates)
#define M_SZ 4096   // B*T

#if __has_builtin(__builtin_amdgcn_global_load_async_to_lds_b128) && \
    __has_builtin(__builtin_amdgcn_s_wait_asynccnt)
#define HAS_ASYNC_LDS 1
#else
#define HAS_ASYNC_LDS 0
#endif

#if __has_builtin(__builtin_amdgcn_tensor_load_to_lds) && \
    __has_builtin(__builtin_amdgcn_s_wait_tensorcnt)
#define HAS_TDM 1
#else
#define HAS_TDM 0
#endif

// ---------------- helpers ----------------

__device__ __forceinline__ bf16_t to_bf16(float f) {
  unsigned u = __float_as_uint(f);
  u += 0x7FFFu + ((u >> 16) & 1u);          // round to nearest even
  unsigned short s = (unsigned short)(u >> 16);
  return __builtin_bit_cast(bf16_t, s);
}

// Stage one 16-byte chunk global -> LDS (async when the HW path is exposed).
__device__ __forceinline__ void stage_b128(const bf16_t* g, bf16_t* l) {
#if HAS_ASYNC_LDS
  __builtin_amdgcn_global_load_async_to_lds_b128(
      (as1_v4i_t)g, (as3_v4i_t)l, 0, 0);
#else
  *reinterpret_cast<uint4*>(l) = *reinterpret_cast<const uint4*>(g);
#endif
}

__device__ __forceinline__ void stage_wait() {
#if HAS_ASYNC_LDS
  __builtin_amdgcn_s_wait_asynccnt(0);
#endif
}

// Load a 16x32 bf16 fragment from a row-major tile (row stride `ld` elements).
// Works for A ([M,K] row-major, lane = M) and B built from row-major [N,K]
// weights (lane = N), per the CDNA5 16-bit matrix VGPR layout:
//   lanes 0-15 : row = lane,    elems 0..7 = K0..7,  elems 8..15 = K16..23
//   lanes 16-31: row = lane-16, elems 0..7 = K8..15, elems 8..15 = K24..31
__device__ __forceinline__ v16bf load_frag_k32(const bf16_t* base, int ld) {
  const int lane = threadIdx.x & 31;
  const int r  = lane & 15;
  const int kh = (lane >> 4) << 3;
  const bf16_t* p = base + (size_t)r * ld + kh;
  v8bf lo = *reinterpret_cast<const v8bf*>(p);
  v8bf hi = *reinterpret_cast<const v8bf*>(p + 16);
  return __builtin_shufflevector(lo, hi, 0, 1, 2, 3, 4, 5, 6, 7,
                                         8, 9, 10, 11, 12, 13, 14, 15);
}

__device__ __forceinline__ v8f wmma_bf16f32(v16bf a, v16bf b, v8f c) {
  return __builtin_amdgcn_wmma_f32_16x16x32_bf16(
      /*neg_a=*/false, a, /*neg_b=*/false, b,
      /*c_mod=*/(short)0, c, /*reuse_a=*/false, /*reuse_b=*/false);
}

// ---------------- setup: biases, zero h-state, zero barriers ----------------

__global__ void setup_kernel(const float* __restrict__ bih0, const float* __restrict__ bhh0,
                             float* __restrict__ bsum0,
                             const float* __restrict__ bih1, const float* __restrict__ bhh1,
                             float* __restrict__ bsum1,
                             bf16_t* __restrict__ h0, bf16_t* __restrict__ h1,
                             unsigned* __restrict__ bars) {
  int i = blockIdx.x * blockDim.x + threadIdx.x;
  if (i < G_SZ) {
    bsum0[i] = bih0[i] + bhh0[i];
    bsum1[i] = bih1[i] + bhh1[i];
  }
  if (i < 2 * B_SZ * H_SZ) {
    bf16_t z = __builtin_bit_cast(bf16_t, (unsigned short)0);
    h0[i] = z;
    h1[i] = z;
  }
  if (i < 64) bars[i] = 0u;
}

// ---------------- f32 -> bf16 conversion ----------------

__global__ void f32_to_bf16_kernel(const float* __restrict__ in,
                                   bf16_t* __restrict__ out, int n) {
  int stride = gridDim.x * blockDim.x;
  for (int i = blockIdx.x * blockDim.x + threadIdx.x; i < n; i += stride)
    out[i] = to_bf16(in[i]);
}

// ---------------- embedding gather (+ bf16 cast) ----------------

__global__ void embed_gather_kernel(const int* __restrict__ x,
                                    const float* __restrict__ embed,
                                    bf16_t* __restrict__ xe) {
  int m = blockIdx.x;                     // m = b*T + t
  int tok = x[m];
  if (tok > V_SZ) tok = V_SZ;             // clamp OOV to padding row
  const float* src = embed + (size_t)tok * E_SZ;
  bf16_t* dst = xe + (size_t)m * E_SZ;
  for (int e = threadIdx.x; e < E_SZ; e += blockDim.x)
    dst[e] = to_bf16(src[e]);
}

// ---------------- unified GEMM:  out = A @ W^T + bias ----------------
// A: [4096, 1024] bf16 row-major, W: [N, 1024] bf16 row-major.
// Block: 256 threads = 8 waves stacked in M. Block tile 128(M) x 128(N).
// The 128x32 B slab is staged cooperatively into double-buffered LDS
// (async-to-LDS), shared by all 8 waves; A fragments are register-prefetched.
// Epilogue: XG_STORE ? scatter into xg[T][B][4H] : row-major [4096, NCOLS].

template <bool XG_STORE, int NCOLS>
__global__ __launch_bounds__(256) void gemm_bf16_kernel(const bf16_t* __restrict__ A,
                                                        const bf16_t* __restrict__ W,
                                                        const float* __restrict__ bias,
                                                        float* __restrict__ out) {
  __shared__ __attribute__((aligned(16))) bf16_t bs[2][128 * 32];

  const int K = H_SZ;
  const int tid  = threadIdx.x;
  const int wave = tid >> 5;
  const int lane = tid & 31;
  const int m0 = blockIdx.y * 128 + wave * 16;
  const int n0 = blockIdx.x * 128;
  const int nl = lane & 15;
  const int mb = (lane >> 4) << 3;

  v8f acc[8];
#pragma unroll
  for (int nt = 0; nt < 8; ++nt) {
    float bv = bias[n0 + nt * 16 + nl];
#pragma unroll
    for (int r = 0; r < 8; ++r) acc[nt][r] = bv;
  }

  // Cooperative stage of one 128(N) x 32(K) bf16 slab: 512 x 16B chunks.
  auto stageB = [&](int buf, int kk) {
#pragma unroll
    for (int i = 0; i < 2; ++i) {
      int c = tid + i * 256;
      int row = c >> 2;                 // 4 chunks per 32-elem row
      int cc  = c & 3;
      stage_b128(W + (size_t)(n0 + row) * K + kk + cc * 8,
                 &bs[buf][row * 32 + cc * 8]);
    }
  };

  stageB(0, 0);
  v16bf af = load_frag_k32(A + (size_t)m0 * K, K);   // A slab for k=0

  for (int kt = 0; kt < K / 32; ++kt) {
    stage_wait();
    __syncthreads();                     // B slab `cur` visible to all waves
    const int cur = kt & 1;
    if (kt + 1 < K / 32) stageB(cur ^ 1, (kt + 1) * 32);

    v16bf af_next = af;
    if (kt + 1 < K / 32)
      af_next = load_frag_k32(A + (size_t)m0 * K + (kt + 1) * 32, K);

#pragma unroll
    for (int nt = 0; nt < 8; ++nt) {
      v16bf bfz = load_frag_k32(&bs[cur][nt * 16 * 32], 32);
      acc[nt] = wmma_bf16f32(af, bfz, acc[nt]);
    }
    af = af_next;
  }

#pragma unroll
  for (int nt = 0; nt < 8; ++nt) {
    int n = n0 + nt * 16 + nl;
#pragma unroll
    for (int r = 0; r < 8; ++r) {
      int m = m0 + mb + r;              // m = b*T + t
      if (XG_STORE) {
        int b = m >> 8;
        int t = m & (T_SZ - 1);
        out[(size_t)((t << 4) + b) * NCOLS + n] = acc[nt][r];
      } else {
        out[(size_t)m * NCOLS + n] = acc[nt][r];
      }
    }
  }
}

// ---------------- persistent recurrent kernel (one layer) ----------------
// 64 workgroups x 128 threads (4 waves). WG g owns hidden units [16g,16g+16).
// Wave w computes gate w's 16x16 tile: gates = xg_t + h_{t-1} @ Whh^T.
// Whh slice (4 gate-blocks x 16 rows x 1024 K bf16 = 128 KB) staged into LDS
// once via the Tensor Data Mover (3-D tile descriptor), async-LDS fallback.
// Device-wide release/acquire spin barrier per timestep; h double-buffered.

__global__ __launch_bounds__(128) void lstm_recurrent_kernel(
    const float* __restrict__ xg, const bf16_t* __restrict__ Whh,
    bf16_t* __restrict__ hstate /* [2][16][1024] */,
    bf16_t* __restrict__ hseq   /* [B*T][1024]   */,
    unsigned* __restrict__ bar) {
  __shared__ __attribute__((aligned(16))) bf16_t whh_s[64 * H_SZ];
  __shared__ float gates_s[4][16][16];
  __shared__ float c_s[16][16];

  const int tid  = threadIdx.x;
  const int wave = tid >> 5;      // gate index 0..3 (i, f, g, o)
  const int lane = tid & 31;
  const int wg   = blockIdx.x;    // 0..63, hidden-unit block

#if HAS_TDM
  // One TDM descriptor moves the whole 3-D tile:
  //   dim0 (X): 1024 K elements (contiguous row)
  //   dim1 (Y): 16 rows, stride 1024 elements
  //   dim2 (Z): 4 gate blocks, stride 1024*1024 elements
  // LDS result is [z*16+y][x] row-major == whh_s local-row layout.
  if (tid < 32) {                 // one wave issues; TDM ignores EXEC
    unsigned lds_base =
        (unsigned)(unsigned long long)(__attribute__((address_space(3))) bf16_t*)whh_s;
    unsigned long long ga =
        (unsigned long long)(const void*)(Whh + (size_t)(wg << 4) * H_SZ);
    u32x4_t g0;
    g0[0] = 1u;                                           // count=1, user D#
    g0[1] = lds_base;                                     // lds_addr (bytes)
    g0[2] = (unsigned)(ga & 0xFFFFFFFFull);               // global_addr[31:0]
    g0[3] = (unsigned)((ga >> 32) & 0x01FFFFFFull)        // global_addr[56:32]
            | (2u << 30);                                 // type = 2 (image)
    i32x8_t g1;
    g1[0] = 0x00010000;   // workgroup_mask=0, data_size=1 (2 bytes)
    g1[1] = 0x04000000;   // atomic_barrier_addr=0, tensor_dim0.lo16 = 1024
    g1[2] = 0x00100000;   // tensor_dim0.hi16 = 0, tensor_dim1.lo16 = 16
    g1[3] = 0x04000000;   // tensor_dim1.hi16 = 0, tile_dim0 = 1024
    g1[4] = 0x00040010;   // tile_dim1 = 16, tile_dim2 = 4
    g1[5] = 1024;         // tensor_dim0_stride[31:0] = 1024
    g1[6] = 0;            // stride0[47:32]=0, stride1[15:0]=0
    g1[7] = 16;           // tensor_dim1_stride[47:16] = 16  (=> 1048576)
    i32x4_t g2;
    g2[0] = 4;            // tensor_dim2 = 4
    g2[1] = 0;            // tensor_dim3 (unused)
    g2[2] = 0;            // tensor_dim2_stride (unused)
    g2[3] = 0;            // tile_dim3 = 0 (unused)
    i32x4_t g3 = {0, 0, 0, 0};
    i32x8_t g4 = {0, 0, 0, 0, 0, 0, 0, 0};   // extra group (6-arg form)
    __builtin_amdgcn_tensor_load_to_lds(g0, g1, g2, g3, g4, 0);
    __builtin_amdgcn_s_wait_tensorcnt(0);
  }
  for (int i = tid; i < 256; i += 128) (&c_s[0][0])[i] = 0.f;
#else
  // Fallback: async-LDS staged copy of this WG's 64 Whh rows.
  for (int i = tid; i < 64 * (H_SZ / 8); i += 128) {
    int lr = i >> 7;                                    // local row 0..63
    int c  = i & 127;                                   // 16B chunk in row
    int grow = ((lr >> 4) << 10) + (wg << 4) + (lr & 15);
    stage_b128(Whh + (size_t)grow * H_SZ + c * 8, &whh_s[lr * H_SZ + c * 8]);
  }
  for (int i = tid; i < 256; i += 128) (&c_s[0][0])[i] = 0.f;
  stage_wait();
#endif
  __syncthreads();

  const int nl = lane & 15;
  const int mb = (lane >> 4) << 3;
  const int gcol = wave * H_SZ + (wg << 4) + nl;          // global gate column
  const bf16_t* bsrc = whh_s + (size_t)(wave * 16) * H_SZ;

  for (int t = 0; t < T_SZ; ++t) {
    const bf16_t* hrd = hstate + (size_t)(t & 1) * (B_SZ * H_SZ);
    bf16_t* hwr = hstate + (size_t)((t + 1) & 1) * (B_SZ * H_SZ);

    // Accumulator initialized with the precomputed input gates xg[t].
    v8f acc;
    {
      const float* xr = xg + (size_t)(t << 4) * G_SZ + gcol;
#pragma unroll
      for (int r = 0; r < 8; ++r) acc[r] = xr[(size_t)(mb + r) * G_SZ];
    }

    // Software-pipelined K loop: prefetch next A (global) / B (LDS) frags.
    v16bf af = load_frag_k32(hrd, H_SZ);
    v16bf bfz = load_frag_k32(bsrc, H_SZ);
    for (int k = 0; k < H_SZ; k += 32) {
      v16bf af_n = af, bf_n = bfz;
      if (k + 32 < H_SZ) {
        af_n = load_frag_k32(hrd + k + 32, H_SZ);
        bf_n = load_frag_k32(bsrc + k + 32, H_SZ);
      }
      acc = wmma_bf16f32(af, bfz, acc);
      af = af_n;
      bfz = bf_n;
    }

#pragma unroll
    for (int r = 0; r < 8; ++r) gates_s[wave][mb + r][nl] = acc[r];
    __syncthreads();

    // Activations + state update: 256 (b, j) pairs over 128 threads.
    for (int p = tid; p < 256; p += 128) {
      int b = p >> 4, jl = p & 15;
      float iv = gates_s[0][b][jl];
      float fv = gates_s[1][b][jl];
      float gv = gates_s[2][b][jl];
      float ov = gates_s[3][b][jl];
      iv = 1.f / (1.f + __expf(-iv));
      fv = 1.f / (1.f + __expf(-fv));
      ov = 1.f / (1.f + __expf(-ov));
      gv = tanhf(gv);
      float cv = fv * c_s[b][jl] + iv * gv;
      c_s[b][jl] = cv;
      float hv = ov * tanhf(cv);
      bf16_t hb = to_bf16(hv);
      int J = (wg << 4) + jl;
      hwr[(size_t)b * H_SZ + J] = hb;
      hseq[(size_t)(b * T_SZ + t) * H_SZ + J] = hb;
    }
    __syncthreads();

    // Device-wide barrier: release our h writes, acquire everyone else's.
    if (tid == 0) {
      __hip_atomic_fetch_add(bar, 1u, __ATOMIC_RELEASE, __HIP_MEMORY_SCOPE_AGENT);
      unsigned target = 64u * (unsigned)(t + 1);
      while (__hip_atomic_load(bar, __ATOMIC_ACQUIRE, __HIP_MEMORY_SCOPE_AGENT) < target) {
        __builtin_amdgcn_s_sleep(2);
      }
    }
    __syncthreads();
  }
}

// ---------------- host launcher ----------------

extern "C" void kernel_launch(void* const* d_in, const int* in_sizes, int n_in,
                              void* d_out, int out_size, void* d_ws, size_t ws_size,
                              hipStream_t stream) {
  (void)in_sizes; (void)n_in; (void)out_size; (void)ws_size;

  const int*   x     = (const int*)  d_in[0];
  const float* embed = (const float*)d_in[1];
  const float* Wih0  = (const float*)d_in[2];
  const float* Whh0  = (const float*)d_in[3];
  const float* bih0  = (const float*)d_in[4];
  const float* bhh0  = (const float*)d_in[5];
  const float* Wih1  = (const float*)d_in[6];
  const float* Whh1  = (const float*)d_in[7];
  const float* bih1  = (const float*)d_in[8];
  const float* bhh1  = (const float*)d_in[9];
  const float* Wlin  = (const float*)d_in[10];
  const float* blin  = (const float*)d_in[11];
  float* out = (float*)d_out;

  char* p = (char*)d_ws;
  auto take = [&](size_t bytes) -> char* {
    char* q = p;
    p += (bytes + 255) & ~((size_t)255);
    return q;
  };

  unsigned* bars = (unsigned*)take(256);
  float* bsum0   = (float*)take((size_t)G_SZ * 4);
  float* bsum1   = (float*)take((size_t)G_SZ * 4);
  bf16_t* hst0   = (bf16_t*)take((size_t)2 * B_SZ * H_SZ * 2);
  bf16_t* hst1   = (bf16_t*)take((size_t)2 * B_SZ * H_SZ * 2);
  bf16_t* xe     = (bf16_t*)take((size_t)M_SZ * E_SZ * 2);
  bf16_t* h1     = (bf16_t*)take((size_t)M_SZ * H_SZ * 2);
  bf16_t* h2     = (bf16_t*)take((size_t)M_SZ * H_SZ * 2);
  bf16_t* wih0b  = (bf16_t*)take((size_t)G_SZ * E_SZ * 2);
  bf16_t* whh0b  = (bf16_t*)take((size_t)G_SZ * H_SZ * 2);
  bf16_t* wih1b  = (bf16_t*)take((size_t)G_SZ * H_SZ * 2);
  bf16_t* whh1b  = (bf16_t*)take((size_t)G_SZ * H_SZ * 2);
  bf16_t* wlinb  = (bf16_t*)take((size_t)V_SZ * H_SZ * 2);
  float*  xg     = (float*)take((size_t)T_SZ * B_SZ * G_SZ * 4);

  setup_kernel<<<128, 256, 0, stream>>>(bih0, bhh0, bsum0, bih1, bhh1, bsum1,
                                        hst0, hst1, bars);
  f32_to_bf16_kernel<<<2048, 256, 0, stream>>>(Wih0, wih0b, G_SZ * E_SZ);
  f32_to_bf16_kernel<<<2048, 256, 0, stream>>>(Whh0, whh0b, G_SZ * H_SZ);
  f32_to_bf16_kernel<<<2048, 256, 0, stream>>>(Wih1, wih1b, G_SZ * H_SZ);
  f32_to_bf16_kernel<<<2048, 256, 0, stream>>>(Whh1, whh1b, G_SZ * H_SZ);
  f32_to_bf16_kernel<<<4096, 256, 0, stream>>>(Wlin, wlinb, V_SZ * H_SZ);
  embed_gather_kernel<<<M_SZ, 256, 0, stream>>>(x, embed, xe);

  // Layer 0
  gemm_bf16_kernel<true, G_SZ><<<dim3(G_SZ / 128, M_SZ / 128), 256, 0, stream>>>(
      xe, wih0b, bsum0, xg);
  lstm_recurrent_kernel<<<64, 128, 0, stream>>>(xg, whh0b, hst0, h1, bars + 0);

  // Layer 1
  gemm_bf16_kernel<true, G_SZ><<<dim3(G_SZ / 128, M_SZ / 128), 256, 0, stream>>>(
      h1, wih1b, bsum1, xg);
  lstm_recurrent_kernel<<<64, 128, 0, stream>>>(xg, whh1b, hst1, h2, bars + 1);

  // Logits
  gemm_bf16_kernel<false, V_SZ><<<dim3(V_SZ / 128, M_SZ / 128), 256, 0, stream>>>(
      h2, wlinb, blin, out);
}
